// GCN_22797686407654
// MI455X (gfx1250) — matrix operand order repaired and checked by reading, not compiled, not run on hardware
//
#include <hip/hip_runtime.h>
#include <math.h>

// ---------------- problem constants (match reference) ----------------
#define NN   20000
#define EE   320000
#define DIN  128
#define DH   64
#define NH   8
#define HD   (NH*DH)    // 512
#define NB   16
#define NC   10
#define NEG_SLOPE 0.2f

typedef _Float16 v16h __attribute__((ext_vector_type(16)));
typedef _Float16 v8h  __attribute__((ext_vector_type(8)));
typedef float    v8f  __attribute__((ext_vector_type(8)));

// ---------------- utility ----------------
__global__ __launch_bounds__(256) void fill_f32(float* __restrict__ p, float v, int n) {
    int i = blockIdx.x * blockDim.x + threadIdx.x;
    if (i < n) p[i] = v;
}

__device__ __forceinline__ void atomicMaxF(float* addr, float val) {
    // standard sign-split trick: works for mixed-sign floats
    if (val >= 0.f) atomicMax((int*)addr, __float_as_int(val));
    else            atomicMin((unsigned int*)addr, __float_as_uint(val));
}

// ---------------- WMMA GEMM: C[M,Nc] = A[M,K] @ B[K,Nc] (+bias) ----------------
// fp32 in / fp32 out, f16 WMMA compute (f32 accumulate).
// K is a compile-time constant (128 or 64): the whole K-panel is staged in LDS
// once, then all K/32 WMMA steps run back-to-back with a single barrier.
// Block = 256 threads = 8 waves; each wave owns a 16x16 tile of C;
// block covers 128 rows x 16 cols. Requires K % 32 == 0, Nc % 16 == 0.
template <int K>
__global__ __launch_bounds__(256)
void gemm_f16_wmma(const float* __restrict__ A, const float* __restrict__ B,
                   const float* __restrict__ bias, float* __restrict__ C,
                   int M, int Nc) {
    __shared__ alignas(16) _Float16 As[128 * K];   // [row][k] panel
    __shared__ alignas(16) _Float16 Bs[16 * K];    // transposed [n][k] panel

    const int tid  = threadIdx.x;
    const int lane = tid & 31;
    const int wave = tid >> 5;
    const int m_block = blockIdx.x * 128;
    const int n0 = blockIdx.y * 16;

    // --- stage A panel: 128 rows x K, fp32 -> f16 (2 threads per row) ---
    {
        int row = tid >> 1;
        int col = (tid & 1) * (K / 2);
        int gm = m_block + row;
        _Float16* dst = &As[row * K + col];
        if (gm < M) {
            const float* src = &A[(size_t)gm * K + col];
            #pragma unroll
            for (int i = 0; i < K / 2; ++i) dst[i] = (_Float16)src[i];
        } else {
            #pragma unroll
            for (int i = 0; i < K / 2; ++i) dst[i] = (_Float16)0.f;
        }
    }
    // --- stage B panel transposed: Bs[n][k] ---
    if (tid < K) {
        const float* src = &B[(size_t)tid * Nc + n0];
        #pragma unroll
        for (int j = 0; j < 16; ++j) Bs[j * K + tid] = (_Float16)src[j];
    }
    __syncthreads();

    // --- compute: fully unrolled chain of K/32 WMMA ops out of LDS ---
    const int mrow = lane & 15;
    const int kbA  = (lane >> 4) * 8;    // A: halves 0-7 -> K kbA.., 8-15 -> K kbA+16..
    const int kbB  = (lane >> 4) * 8;    // B: halves 0-7 -> K kbB.., 8-15 -> K kbB+16..
    const _Float16* arow = &As[(wave * 16 + mrow) * K];
    const _Float16* brow = &Bs[(lane & 15) * K];

    v8f acc = {};
    #pragma unroll
    for (int kt = 0; kt < K; kt += 32) {
        v8h alo = *(const v8h*)(arow + kt + kbA);
        v8h ahi = *(const v8h*)(arow + kt + kbA + 16);
        v16h a;
        #pragma unroll
        for (int i = 0; i < 8; ++i) { a[i] = alo[i]; a[i + 8] = ahi[i]; }

        v8h blo = *(const v8h*)(brow + kt + kbB);
        v8h bhi = *(const v8h*)(brow + kt + kbB + 16);
        v16h b;
        #pragma unroll
        for (int i = 0; i < 8; ++i) { b[i] = blo[i]; b[i + 8] = bhi[i]; }

        acc = __builtin_amdgcn_wmma_f32_16x16x32_f16(
                  false, a, false, b, (short)0, acc, false, false);
    }

    // --- epilogue: D layout row = (lane>>4)*8 + r, col = lane&15 ---
    const int col = n0 + (lane & 15);
    const int mbase = m_block + wave * 16 + (lane >> 4) * 8;
    const float bv = bias ? bias[col] : 0.f;
    #pragma unroll
    for (int r = 0; r < 8; ++r) {
        int gm = mbase + r;
        if (gm < M) C[(size_t)gm * Nc + col] = acc[r] + bv;
    }
}

// ---------------- el/er: per (node, head) dot with attn vectors ----------------
__global__ __launch_bounds__(256)
void node_elr(const float* __restrict__ feat, const float* __restrict__ al,
              const float* __restrict__ ar, float* __restrict__ el,
              float* __restrict__ er) {
    int i = blockIdx.x * blockDim.x + threadIdx.x;   // i = n*NH + h
    if (i >= NN * NH) return;
    int h = i & (NH - 1);
    const float* f = feat + (size_t)i * DH;          // (n*NH+h)*DH == n*HD + h*DH
    const float* L = al + h * DH;
    const float* R = ar + h * DH;
    float sl = 0.f, sr = 0.f;
    #pragma unroll 8
    for (int d = 0; d < DH; ++d) { sl += f[d] * L[d]; sr += f[d] * R[d]; }
    el[i] = sl;
    er[i] = sr;
}

// ---------------- edge pass 1: leaky_relu score + segment max ----------------
__global__ __launch_bounds__(256)
void edge_score_max(const int* __restrict__ src, const int* __restrict__ dst,
                    const float* __restrict__ el, const float* __restrict__ er,
                    float* __restrict__ esc, float* __restrict__ mmax) {
    int i = blockIdx.x * blockDim.x + threadIdx.x;   // i = e*NH + h
    if (i >= EE * NH) return;
    int e = i >> 3, h = i & (NH - 1);
    float x = el[src[e] * NH + h] + er[dst[e] * NH + h];
    x = (x > 0.f) ? x : NEG_SLOPE * x;
    esc[i] = x;
    atomicMaxF(&mmax[dst[e] * NH + h], x);
}

// ---------------- edge pass 2: exp and segment sum ----------------
__global__ __launch_bounds__(256)
void edge_expsum(const int* __restrict__ dst, float* __restrict__ esc,
                 const float* __restrict__ mmax, float* __restrict__ den) {
    int i = blockIdx.x * blockDim.x + threadIdx.x;
    if (i >= EE * NH) return;
    int e = i >> 3, h = i & (NH - 1);
    float ex = __expf(esc[i] - mmax[dst[e] * NH + h]);
    esc[i] = ex;
    atomicAdd(&den[dst[e] * NH + h], ex);
}

// ---------------- edge pass 3: normalize alpha ----------------
__global__ __launch_bounds__(256)
void edge_norm(const int* __restrict__ dst, float* __restrict__ esc,
               const float* __restrict__ den) {
    int i = blockIdx.x * blockDim.x + threadIdx.x;
    if (i >= EE * NH) return;
    int e = i >> 3, h = i & (NH - 1);
    esc[i] = esc[i] / den[dst[e] * NH + h];
}

// ---------------- edge pass 4: wave-per-edge weighted gather/scatter ----------------
__global__ __launch_bounds__(256)
void edge_aggregate(const int* __restrict__ src, const int* __restrict__ dst,
                    const float* __restrict__ feat, const float* __restrict__ alpha,
                    float* __restrict__ outp) {
    int wid  = (blockIdx.x * blockDim.x + threadIdx.x) >> 5;
    int lane = threadIdx.x & 31;
    int nwaves = (gridDim.x * blockDim.x) >> 5;
    for (int e = wid; e < EE; e += nwaves) {
        int s = src[e], d = dst[e];
        // prefetch the next edge's 2KB source row: 1 byte/lane, 64B stride
        // -> the 32 lanes cover the whole row (emits global_prefetch_b8)
        if (e + nwaves < EE) {
            int s2 = src[e + nwaves];
            __builtin_prefetch(feat + (size_t)s2 * HD + lane * 16, 0, 0);
        }
        const float* fs = feat + (size_t)s * HD;
        float* od = outp + (size_t)d * HD;
        const float* al = alpha + (size_t)e * NH;
        #pragma unroll
        for (int i = 0; i < HD / 32; ++i) {
            int idx = lane + (i << 5);
            float a = al[idx >> 6];               // head = idx / DH
            atomicAdd(&od[idx], fs[idx] * a);
        }
    }
}

// ---------------- head mean (+ bias, optional relu) ----------------
__global__ __launch_bounds__(256)
void head_mean(const float* __restrict__ agg, const float* __restrict__ bias,
               float* __restrict__ hout, int do_relu) {
    int i = blockIdx.x * blockDim.x + threadIdx.x;   // i = n*DH + d
    if (i >= NN * DH) return;
    int n = i >> 6, d = i & (DH - 1);
    float s = 0.f;
    #pragma unroll
    for (int h = 0; h < NH; ++h) s += agg[(size_t)n * HD + h * DH + d] + bias[h * DH + d];
    s *= (1.0f / NH);
    if (do_relu) s = fmaxf(s, 0.f);
    hout[i] = s;
}

// ---------------- per-graph mean readout (accumulate) ----------------
__global__ __launch_bounds__(256)
void readout(const float* __restrict__ h, const int* __restrict__ gid,
             float* __restrict__ sums, float* __restrict__ cnt) {
    int i = blockIdx.x * blockDim.x + threadIdx.x;   // i = n*DH + d
    if (i >= NN * DH) return;
    int n = i >> 6, d = i & (DH - 1);
    int g = gid[n];
    atomicAdd(&sums[g * DH + d], h[i]);
    if (d == 0) atomicAdd(&cnt[g], 1.0f);
}

// ---------------- classifier head + softmax ----------------
__global__ __launch_bounds__(32)
void head_softmax(const float* __restrict__ sums, const float* __restrict__ cnt,
                  const float* __restrict__ Wh, const float* __restrict__ bh,
                  float* __restrict__ outp) {
    int b = threadIdx.x;
    if (b >= NB) return;
    float inv = 1.0f / fmaxf(cnt[b], 1.0f);
    float logits[NC];
    float mx = -1e30f;
    for (int c = 0; c < NC; ++c) {
        float s = bh[c];
        for (int d = 0; d < DH; ++d) s += sums[b * DH + d] * inv * Wh[d * NC + c];
        logits[c] = s;
        mx = fmaxf(mx, s);
    }
    float dn = 0.f;
    for (int c = 0; c < NC; ++c) { logits[c] = __expf(logits[c] - mx); dn += logits[c]; }
    for (int c = 0; c < NC; ++c) outp[b * NC + c] = logits[c] / dn;
}

// ---------------- launcher ----------------
extern "C" void kernel_launch(void* const* d_in, const int* in_sizes, int n_in,
                              void* d_out, int out_size, void* d_ws, size_t ws_size,
                              hipStream_t stream) {
    const float* g_feats  = (const float*)d_in[0];
    const int*   edge_src = (const int*)  d_in[1];
    const int*   edge_dst = (const int*)  d_in[2];
    const int*   ngid     = (const int*)  d_in[3];
    const float* W_in     = (const float*)d_in[4];
    const float* b_in     = (const float*)d_in[5];
    const float* W1       = (const float*)d_in[6];
    const float* attn_l1  = (const float*)d_in[7];
    const float* attn_r1  = (const float*)d_in[8];
    const float* bias1    = (const float*)d_in[9];
    const float* W2       = (const float*)d_in[10];
    const float* attn_l2  = (const float*)d_in[11];
    const float* attn_r2  = (const float*)d_in[12];
    const float* bias2    = (const float*)d_in[13];
    const float* W_head   = (const float*)d_in[14];
    const float* b_head   = (const float*)d_in[15];
    float* outp = (float*)d_out;

    // workspace carve (~100 MB of f32)
    float* ws   = (float*)d_ws;
    float* feat = ws;  ws += (size_t)NN * HD;   // [N, H*DH]
    float* agg  = ws;  ws += (size_t)NN * HD;   // aggregation accumulator
    float* hbuf = ws;  ws += (size_t)NN * DH;   // node features between layers
    float* el   = ws;  ws += (size_t)NN * NH;
    float* er   = ws;  ws += (size_t)NN * NH;
    float* mm   = ws;  ws += (size_t)NN * NH;
    float* den  = ws;  ws += (size_t)NN * NH;
    float* esc  = ws;  ws += (size_t)EE * NH;   // edge scores / alphas
    float* sums = ws;  ws += (size_t)NB * DH;
    float* cnt  = ws;  ws += (size_t)NB;

    const int T = 256;
    #define GRID1(n) dim3(((n) + T - 1) / T)

    // ---- input layer: hbuf = g_feats @ W_in + b_in  (K = 128) ----
    {
        dim3 g((NN + 127) / 128, DH / 16);
        gemm_f16_wmma<DIN><<<g, T, 0, stream>>>(g_feats, W_in, b_in, hbuf, NN, DH);
    }

    for (int layer = 0; layer < 2; ++layer) {
        const float* Wl    = layer == 0 ? W1      : W2;
        const float* aL    = layer == 0 ? attn_l1 : attn_l2;
        const float* aR    = layer == 0 ? attn_r1 : attn_r2;
        const float* biasl = layer == 0 ? bias1   : bias2;
        int do_relu        = layer == 0 ? 1 : 0;

        // feat = hbuf @ Wl   (N x 64) @ (64 x 512)   (K = 64)
        {
            dim3 g((NN + 127) / 128, HD / 16);
            gemm_f16_wmma<DH><<<g, T, 0, stream>>>(hbuf, Wl, nullptr, feat, NN, HD);
        }
        node_elr<<<GRID1(NN * NH), T, 0, stream>>>(feat, aL, aR, el, er);

        fill_f32<<<GRID1(NN * NH), T, 0, stream>>>(mm,  -1e30f, NN * NH);
        fill_f32<<<GRID1(NN * NH), T, 0, stream>>>(den,  0.f,   NN * NH);
        fill_f32<<<GRID1(NN * HD), T, 0, stream>>>(agg,  0.f,   NN * HD);

        edge_score_max<<<GRID1(EE * NH), T, 0, stream>>>(edge_src, edge_dst, el, er, esc, mm);
        edge_expsum   <<<GRID1(EE * NH), T, 0, stream>>>(edge_dst, esc, mm, den);
        edge_norm     <<<GRID1(EE * NH), T, 0, stream>>>(edge_dst, esc, den);
        // one wave per edge: 8 waves/block
        edge_aggregate<<<dim3((EE + 7) / 8), T, 0, stream>>>(edge_src, edge_dst, feat, esc, agg);

        head_mean<<<GRID1(NN * DH), T, 0, stream>>>(agg, biasl, hbuf, do_relu);
    }

    // ---- graph mean readout + classifier + softmax ----
    fill_f32<<<GRID1(NB * DH), T, 0, stream>>>(sums, 0.f, NB * DH);
    fill_f32<<<dim3(1), T, 0, stream>>>(cnt, 0.f, NB);
    readout<<<GRID1(NN * DH), T, 0, stream>>>(hbuf, ngid, sums, cnt);
    head_softmax<<<dim3(1), 32, 0, stream>>>(sums, cnt, W_head, b_head, outp);

    (void)in_sizes; (void)n_in; (void)out_size; (void)ws_size;
}
